// MoEFFN_2164663517568
// MI455X (gfx1250) — matrix operand order repaired and checked by reading, not compiled, not run on hardware
//
#include <hip/hip_runtime.h>
#include <hip/hip_bf16.h>

// ---------------------------------------------------------------------------
// MoE FFN (top-2 of 16 experts), DIM=1024, HID=2048, N=8192 tokens, fp32 in/out.
// Router -> compact per-expert token lists -> bf16 WMMA grouped GEMMs:
//   gemm1: h = silu(x @ W1[e] + b1[e])          (WMMA bf16, fp32 accum)
//   gemm2: out[token] += w * (h @ W2[e] + b2[e]) (WMMA + f32 global atomics)
// Weights converted fp32->bf16 and transposed (N-major, K-innermost) so B
// fragments are single contiguous 32B loads per lane; all bf16 weights
// (128MB) stay L2-resident (192MB). M-tile = 32 so each B fragment feeds two
// WMMAs (halves load traffic per matrix op).
// ---------------------------------------------------------------------------

typedef __bf16 bf16;
typedef bf16  v16bf __attribute__((ext_vector_type(16)));
typedef bf16  v8bf  __attribute__((ext_vector_type(8)));
typedef float v8f   __attribute__((ext_vector_type(8)));

#define DIM_      1024
#define HID_      2048
#define EXPERTS_  16
#define NTOK_     8192
#define MT_       32          // M-tile (gathered rows per block), 2 WMMA subtiles
#define KSTEP_    32          // WMMA K per instruction (bf16)
#define NW_       4           // 16-wide N subtiles per wave (A-frag reuse x4)
#define TILES_PER_EXPERT_ 256 // worst case 8192 tokens / MT_
#define TILE_SHIFT_ 8
#define RT_INV_   10.0f       // 1 / ROUTER_TEMP
#define AUXC_     1e-5f

__device__ __forceinline__ unsigned bfbits(float f) {
  unsigned u = __builtin_bit_cast(unsigned, f);
  return (u + 0x7FFFu + ((u >> 16) & 1u)) >> 16;       // round-nearest-even
}
__device__ __forceinline__ bf16 f2bf(float f) {
  unsigned short s = (unsigned short)bfbits(f);
  return __builtin_bit_cast(bf16, s);
}

// ------------------------- x: fp32 -> bf16 (packed stores) -------------------
__global__ __launch_bounds__(256) void cvt_x_kernel(const float* __restrict__ x,
                                                    bf16* __restrict__ xb) {
  size_t i = ((size_t)blockIdx.x * 256 + threadIdx.x) * 4;
  float4 v = *(const float4*)(x + i);
  uint2 p;
  p.x = bfbits(v.x) | (bfbits(v.y) << 16);
  p.y = bfbits(v.z) | (bfbits(v.w) << 16);
  *(uint2*)(xb + i) = p;
}

// ------------------------- router: one wave per token -----------------------
__global__ __launch_bounds__(256) void router_kernel(
    const float* __restrict__ x, const float* __restrict__ Wr,
    const float* __restrict__ br, int2* __restrict__ te, float2* __restrict__ tw,
    int* __restrict__ cnt, float* __restrict__ probsum) {
  __shared__ float psum[EXPERTS_];
  const int lane = threadIdx.x & 31;
  const int wid  = threadIdx.x >> 5;
  if (threadIdx.x < EXPERTS_) psum[threadIdx.x] = 0.f;
  __syncthreads();

  const int n = blockIdx.x * 8 + wid;
  float acc[EXPERTS_];
#pragma unroll
  for (int e = 0; e < EXPERTS_; ++e) acc[e] = 0.f;
  const float* xr = x + (size_t)n * DIM_;
  for (int j = 0; j < DIM_ / 32; ++j) {
    int k = j * 32 + lane;                 // coalesced over lanes
    float xv = xr[k];
    const float* wr = Wr + (size_t)k * EXPERTS_;
#pragma unroll
    for (int e = 0; e < EXPERTS_; ++e) acc[e] = fmaf(xv, wr[e], acc[e]);
  }
#pragma unroll
  for (int e = 0; e < EXPERTS_; ++e) {
#pragma unroll
    for (int off = 16; off >= 1; off >>= 1) acc[e] += __shfl_xor(acc[e], off, 32);
  }
  // all lanes hold full logits; compute top-2 + softmax redundantly
  float v0 = -__builtin_inff(), v1 = -__builtin_inff();
  int e0 = 0, e1 = 0;
#pragma unroll
  for (int e = 0; e < EXPERTS_; ++e) {
    float lg = (acc[e] + br[e]) * RT_INV_;
    acc[e] = lg;
    if (lg > v0)      { v1 = v0; e1 = e0; v0 = lg; e0 = e; }
    else if (lg > v1) { v1 = lg; e1 = e; }
  }
  float t  = __expf(v1 - v0);
  float w0 = 1.f / (1.f + t);
  float w1 = t * w0;
  float ssum = 0.f;
#pragma unroll
  for (int e = 0; e < EXPERTS_; ++e) { acc[e] = __expf(acc[e] - v0); ssum += acc[e]; }
  float inv = 1.f / ssum;
  if (lane == 0) {
    te[n] = make_int2(e0, e1);
    tw[n] = make_float2(w0, w1);
    atomicAdd(&cnt[e0], 1);
    atomicAdd(&cnt[e1], 1);
#pragma unroll
    for (int e = 0; e < EXPERTS_; ++e) atomicAdd(&psum[e], acc[e] * inv);
  }
  __syncthreads();
  if (threadIdx.x < EXPERTS_) atomicAdd(&probsum[threadIdx.x], psum[threadIdx.x]);
}

// ------------------- prefix offsets + aux-loss finalize ----------------------
__global__ void offsets_aux_kernel(const int* __restrict__ cnt, int* __restrict__ off,
                                   int* __restrict__ cnt2,
                                   const float* __restrict__ probsum,
                                   float* __restrict__ aux_out) {
  if (threadIdx.x == 0) {
    int a = 0; float s = 0.f;
    for (int e = 0; e < EXPERTS_; ++e) {
      off[e] = a; a += cnt[e];
      float p = probsum[e]; s += p * p;
    }
    aux_out[0] = s / (float)EXPERTS_ * AUXC_;
  }
  if (threadIdx.x < EXPERTS_) cnt2[threadIdx.x] = 0;
}

// ------------------- build compact per-expert entry lists --------------------
__global__ void build_kernel(const int2* __restrict__ te, const float2* __restrict__ tw,
                             const int* __restrict__ off, int* __restrict__ cnt2,
                             int* __restrict__ etok, float* __restrict__ ew) {
  int n = blockIdx.x * blockDim.x + threadIdx.x;
  if (n >= NTOK_) return;
  int2 e = te[n]; float2 w = tw[n];
  int p0 = atomicAdd(&cnt2[e.x], 1); etok[off[e.x] + p0] = n; ew[off[e.x] + p0] = w.x;
  int p1 = atomicAdd(&cnt2[e.y], 1); etok[off[e.y] + p1] = n; ew[off[e.y] + p1] = w.y;
}

// --------------- fp32 [E][R][C] -> bf16 transposed [E][C][R] -----------------
__global__ __launch_bounds__(256) void transpose_cvt_kernel(
    const float* __restrict__ W, bf16* __restrict__ WT, int R, int C) {
  __shared__ float tile[32][33];
  const int e  = blockIdx.z;
  const int c0 = blockIdx.x * 32, r0 = blockIdx.y * 32;
  const int tx = threadIdx.x & 31, ty = threadIdx.x >> 5;
  const float* Wp = W + (size_t)e * R * C;
  bf16* WTp = WT + (size_t)e * R * C;
#pragma unroll
  for (int i = 0; i < 4; ++i) {
    int r = ty + i * 8;
    tile[r][tx] = Wp[(size_t)(r0 + r) * C + (c0 + tx)];
  }
  __syncthreads();
#pragma unroll
  for (int i = 0; i < 4; ++i) {
    int r = ty + i * 8;
    WTp[(size_t)(c0 + r) * R + (r0 + tx)] = f2bf(tile[tx][r]);
  }
}

__device__ __forceinline__ v16bf afrag(const bf16* row, int k0, int lg) {
  v8bf lo = *(const v8bf*)(row + k0 + 16 * lg);
  v8bf hi = *(const v8bf*)(row + k0 + 16 * lg + 8);
  return __builtin_shufflevector(lo, hi, 0,1,2,3,4,5,6,7,8,9,10,11,12,13,14,15);
}

// --------------------------- GEMM1: h = silu(x W1 + b1) ----------------------
__global__ __launch_bounds__(256) void gemm1_kernel(
    const bf16* __restrict__ xb, const bf16* __restrict__ W1T,
    const float* __restrict__ b1, const int* __restrict__ cnt,
    const int* __restrict__ off, const int* __restrict__ etok,
    bf16* __restrict__ hbuf) {
  const int e  = blockIdx.x >> TILE_SHIFT_;
  const int mt = blockIdx.x & (TILES_PER_EXPERT_ - 1);
  const int cn = cnt[e];
  if (mt * MT_ >= cn) return;
  const int base = off[e] + mt * MT_;
  const int Mv   = min(MT_, cn - mt * MT_);

  __shared__ int stok[MT_];
  if (threadIdx.x < MT_) {
    int m = threadIdx.x;
    stok[m] = etok[base + (m < Mv ? m : 0)];
  }
  __syncthreads();

  const int lane = threadIdx.x & 31;
  const int wid  = threadIdx.x >> 5;
  const int lg   = lane >> 4;              // K-split lane group
  const int lm   = lane & 15;
  const bf16* a0row = xb + (size_t)stok[lm] * DIM_;       // M rows 0..15
  const bf16* a1row = xb + (size_t)stok[16 + lm] * DIM_;  // M rows 16..31
  const bf16* Bbase = W1T + (size_t)e * HID_ * DIM_;

  for (int np = 0; np < HID_ / 512; ++np) {               // 4 passes of 512 cols
    const int ncol0 = np * 512 + wid * (NW_ * 16);
    v8f acc0[NW_], acc1[NW_];
#pragma unroll
    for (int i = 0; i < NW_; ++i) {
      acc0[i] = v8f{0.f,0.f,0.f,0.f,0.f,0.f,0.f,0.f};
      acc1[i] = v8f{0.f,0.f,0.f,0.f,0.f,0.f,0.f,0.f};
    }
    for (int k0 = 0; k0 < DIM_; k0 += KSTEP_) {
      v16bf a0 = afrag(a0row, k0, lg);
      v16bf a1 = afrag(a1row, k0, lg);
#pragma unroll
      for (int i = 0; i < NW_; ++i) {
        const bf16* bp = Bbase + (size_t)(ncol0 + i * 16 + lm) * DIM_ + k0 + 16 * lg;
        v16bf b = *(const v16bf*)bp;                      // contiguous 32B per lane
        acc0[i] = __builtin_amdgcn_wmma_f32_16x16x32_bf16(
            false, a0, false, b, (short)0, acc0[i], false, false);
        acc1[i] = __builtin_amdgcn_wmma_f32_16x16x32_bf16(
            false, a1, false, b, (short)0, acc1[i], false, false);
      }
    }
    // epilogue: bias + silu -> bf16 h
#pragma unroll
    for (int i = 0; i < NW_; ++i) {
      int nc = ncol0 + i * 16 + lm;
      float bias = b1[e * HID_ + nc];
#pragma unroll
      for (int j = 0; j < 8; ++j) {
        int m = j + 8 * lg;
        if (m < Mv) {
          float v = acc0[i][j] + bias;
          hbuf[(size_t)(base + m) * HID_ + nc] = f2bf(v / (1.f + __expf(-v)));
        }
        int m2 = m + 16;
        if (m2 < Mv) {
          float v = acc1[i][j] + bias;
          hbuf[(size_t)(base + m2) * HID_ + nc] = f2bf(v / (1.f + __expf(-v)));
        }
      }
    }
  }
}

// ------------------ GEMM2: out[token] += w * (h W2 + b2) ---------------------
__global__ __launch_bounds__(256) void gemm2_kernel(
    const bf16* __restrict__ hbuf, const bf16* __restrict__ W2T,
    const float* __restrict__ b2, const int* __restrict__ cnt,
    const int* __restrict__ off, const int* __restrict__ etok,
    const float* __restrict__ ew, float* __restrict__ out) {
  const int e  = blockIdx.x >> TILE_SHIFT_;
  const int mt = blockIdx.x & (TILES_PER_EXPERT_ - 1);
  const int cn = cnt[e];
  if (mt * MT_ >= cn) return;
  const int base = off[e] + mt * MT_;
  const int Mv   = min(MT_, cn - mt * MT_);

  __shared__ int   stok[MT_];
  __shared__ float sw[MT_];
  if (threadIdx.x < MT_) {
    int m = threadIdx.x;
    bool valid = m < Mv;
    stok[m] = valid ? etok[base + m] : 0;
    sw[m]   = valid ? ew[base + m]   : 0.f;
  }
  __syncthreads();

  const int lane = threadIdx.x & 31;
  const int wid  = threadIdx.x >> 5;
  const int lg   = lane >> 4;
  const int lm   = lane & 15;
  const bf16* a0row = hbuf + (size_t)(base + lm) * HID_;       // rows 0..15
  const bf16* a1row = hbuf + (size_t)(base + 16 + lm) * HID_;  // rows 16..31
  const bf16* Bbase = W2T + (size_t)e * DIM_ * HID_;

  for (int np = 0; np < DIM_ / 512; ++np) {                    // 2 passes
    const int ncol0 = np * 512 + wid * (NW_ * 16);
    v8f acc0[NW_], acc1[NW_];
#pragma unroll
    for (int i = 0; i < NW_; ++i) {
      acc0[i] = v8f{0.f,0.f,0.f,0.f,0.f,0.f,0.f,0.f};
      acc1[i] = v8f{0.f,0.f,0.f,0.f,0.f,0.f,0.f,0.f};
    }
    for (int k0 = 0; k0 < HID_; k0 += KSTEP_) {
      v16bf a0 = afrag(a0row, k0, lg);
      v16bf a1 = afrag(a1row, k0, lg);
#pragma unroll
      for (int i = 0; i < NW_; ++i) {
        const bf16* bp = Bbase + (size_t)(ncol0 + i * 16 + lm) * HID_ + k0 + 16 * lg;
        v16bf b = *(const v16bf*)bp;
        acc0[i] = __builtin_amdgcn_wmma_f32_16x16x32_bf16(
            false, a0, false, b, (short)0, acc0[i], false, false);
        acc1[i] = __builtin_amdgcn_wmma_f32_16x16x32_bf16(
            false, a1, false, b, (short)0, acc1[i], false, false);
      }
    }
    // epilogue: (acc + b2) * combine_weight, scatter with f32 atomics
    // (each out element gets exactly 2 contributions -> order-invariant sum)
#pragma unroll
    for (int i = 0; i < NW_; ++i) {
      int nc = ncol0 + i * 16 + lm;
      float bias = b2[e * DIM_ + nc];
#pragma unroll
      for (int j = 0; j < 8; ++j) {
        int m = j + 8 * lg;
        float w = sw[m];
        if (w != 0.f)
          atomicAdd(&out[(size_t)stok[m] * DIM_ + nc], (acc0[i][j] + bias) * w);
        int m2 = m + 16;
        float w2 = sw[m2];
        if (w2 != 0.f)
          atomicAdd(&out[(size_t)stok[m2] * DIM_ + nc], (acc1[i][j] + bias) * w2);
      }
    }
  }
}

// ----------------------------------------------------------------------------
extern "C" void kernel_launch(void* const* d_in, const int* in_sizes, int n_in,
                              void* d_out, int out_size, void* d_ws, size_t ws_size,
                              hipStream_t stream) {
  const float* x  = (const float*)d_in[0];
  const float* Wr = (const float*)d_in[1];
  const float* br = (const float*)d_in[2];
  const float* W1 = (const float*)d_in[3];
  const float* b1 = (const float*)d_in[4];
  const float* W2 = (const float*)d_in[5];
  const float* b2 = (const float*)d_in[6];
  float* out = (float*)d_out;                       // [8192*1024] + aux at end

  char* ws = (char*)d_ws;
  size_t o = 0;
  auto alloc = [&](size_t bytes) { size_t r = o; o += (bytes + 255) & ~(size_t)255; return r; };
  bf16*  W1T  = (bf16*)(ws + alloc((size_t)EXPERTS_ * DIM_ * HID_ * 2));
  bf16*  W2T  = (bf16*)(ws + alloc((size_t)EXPERTS_ * DIM_ * HID_ * 2));
  bf16*  xb   = (bf16*)(ws + alloc((size_t)NTOK_ * DIM_ * 2));
  bf16*  hbuf = (bf16*)(ws + alloc((size_t)(2 * NTOK_ + MT_) * HID_ * 2));
  int2*  te   = (int2*)(ws + alloc((size_t)NTOK_ * sizeof(int2)));
  float2* tw  = (float2*)(ws + alloc((size_t)NTOK_ * sizeof(float2)));
  int*   etok = (int*)(ws + alloc((size_t)(2 * NTOK_ + MT_) * 4));
  float* ew   = (float*)(ws + alloc((size_t)(2 * NTOK_ + MT_) * 4));
  int*   cnt  = (int*)(ws + alloc(EXPERTS_ * 4));
  int*   cnt2 = (int*)(ws + alloc(EXPERTS_ * 4));
  int*   off  = (int*)(ws + alloc(EXPERTS_ * 4));
  float* psum = (float*)(ws + alloc(EXPERTS_ * 4));

  // zero accumulators (capturable async memsets)
  hipMemsetAsync(out, 0, (size_t)NTOK_ * DIM_ * sizeof(float), stream);
  hipMemsetAsync(cnt, 0, EXPERTS_ * 4, stream);
  hipMemsetAsync(psum, 0, EXPERTS_ * 4, stream);

  // x fp32 -> bf16
  cvt_x_kernel<<<(NTOK_ * DIM_) / (256 * 4), 256, 0, stream>>>(x, xb);

  // weight convert+transpose: W1 [E][1024][2048] -> [E][2048][1024] bf16, etc.
  transpose_cvt_kernel<<<dim3(HID_ / 32, DIM_ / 32, EXPERTS_), 256, 0, stream>>>(W1, W1T, DIM_, HID_);
  transpose_cvt_kernel<<<dim3(DIM_ / 32, HID_ / 32, EXPERTS_), 256, 0, stream>>>(W2, W2T, HID_, DIM_);

  router_kernel<<<NTOK_ / 8, 256, 0, stream>>>(x, Wr, br, te, tw, cnt, psum);
  offsets_aux_kernel<<<1, 32, 0, stream>>>(cnt, off, cnt2, psum,
                                           out + (size_t)NTOK_ * DIM_);
  build_kernel<<<NTOK_ / 256, 256, 0, stream>>>(te, tw, off, cnt2, etok, ew);

  gemm1_kernel<<<EXPERTS_ * TILES_PER_EXPERT_, 256, 0, stream>>>(
      xb, W1T, b1, cnt, off, etok, hbuf);
  gemm2_kernel<<<EXPERTS_ * TILES_PER_EXPERT_, 256, 0, stream>>>(
      hbuf, W2T, b2, cnt, off, etok, ew, out);
}
